// SelfAttention_57578331570614
// MI455X (gfx1250) — matrix operand order repaired and checked by reading, not compiled, not run on hardware
//
#include <hip/hip_runtime.h>

typedef __bf16 bf16;
typedef __attribute__((ext_vector_type(16))) __bf16 v16bf;
typedef __attribute__((ext_vector_type(8)))  __bf16 v8bf;
typedef __attribute__((ext_vector_type(8)))  float  v8f;
typedef unsigned long long ull;

#define MDIM 1024
#define SEQL 2048
#define NBATCH 4

__device__ __forceinline__ v16bf make_afrag(const bf16* ap) {
    // lanes 0-15: K[0..15]; lanes 16-31: K[16..31]
    v8bf lo = *(const v8bf*)(ap);
    v8bf hi8 = *(const v8bf*)(ap + 8);
    v16bf a;
#pragma unroll
    for (int i = 0; i < 8; ++i) { a[i] = lo[i]; a[i + 8] = hi8[i]; }
    return a;
}

__device__ __forceinline__ v8bf cvt8(const float* p) {
    float4 f0 = *(const float4*)(p);
    float4 f1 = *(const float4*)(p + 4);
    v8bf r;
    r[0] = (bf16)f0.x; r[1] = (bf16)f0.y; r[2] = (bf16)f0.z; r[3] = (bf16)f0.w;
    r[4] = (bf16)f1.x; r[5] = (bf16)f1.y; r[6] = (bf16)f1.z; r[7] = (bf16)f1.w;
    return r;
}

// ---------------------------------------------------------------------------
// Generic bf16 WMMA GEMM:  C[m,n] = scale * sum_k A[m,k] * W[n,k]
//   A: row-major [M x K] bf16, W: row-major [N x K] bf16 (B column-major)
// 128x128 block tile, 256 threads = 8 waves; wave owns 32x64 (8 WMMA/K-step).
// Double-buffered LDS, register-staged pipeline, 1 barrier per K-step.
// ---------------------------------------------------------------------------
__global__ __launch_bounds__(256) void wmma_gemm_bf16(
    const bf16* __restrict__ A, int lda, ull batchA,
    const bf16* __restrict__ W, int ldb, ull batchB,
    float* __restrict__ C, int ldc, ull batchC,
    int K, float scale)
{
    __shared__ __attribute__((aligned(64))) bf16 As[2][128 * 32];
    __shared__ __attribute__((aligned(64))) bf16 Bs[2][128 * 32];

    const int tid  = threadIdx.x;
    const int wave = tid >> 5;
    const int lane = tid & 31;
    const int l15  = lane & 15;
    const int hi   = lane >> 4;
    const int wr   = (wave & 3) * 32;    // wave row offset within tile
    const int wc   = (wave >> 2) * 64;   // wave col offset within tile
    const int m0   = blockIdx.x * 128;
    const int n0   = blockIdx.y * 128;

    A += (ull)blockIdx.z * batchA;
    W += (ull)blockIdx.z * batchB;
    C += (ull)blockIdx.z * batchC;

    // staging: 512 chunks of 16B per 8KB tile, 2 chunks per thread
    const int c0 = tid, c1 = tid + 256;
    const int ra0 = c0 >> 2, ka0 = (c0 & 3) << 3;
    const int ra1 = c1 >> 2, ka1 = (c1 & 3) << 3;
    const int lofs0 = ra0 * 32 + ka0;
    const int lofs1 = ra1 * 32 + ka1;
    const bf16* gA0 = A + (ull)(m0 + ra0) * lda + ka0;
    const bf16* gA1 = A + (ull)(m0 + ra1) * lda + ka1;
    const bf16* gB0 = W + (ull)(n0 + ra0) * ldb + ka0;
    const bf16* gB1 = W + (ull)(n0 + ra1) * ldb + ka1;

    const v8f vzero = {0.f,0.f,0.f,0.f,0.f,0.f,0.f,0.f};
    v8f acc[2][4];
#pragma unroll
    for (int i = 0; i < 2; ++i)
#pragma unroll
        for (int t = 0; t < 4; ++t) acc[i][t] = vzero;

    // prologue: stage k=0 through registers into LDS buffer 0
    v8bf sa0 = *(const v8bf*)gA0;
    v8bf sa1 = *(const v8bf*)gA1;
    v8bf sb0 = *(const v8bf*)gB0;
    v8bf sb1 = *(const v8bf*)gB1;
    *(v8bf*)(&As[0][lofs0]) = sa0;
    *(v8bf*)(&As[0][lofs1]) = sa1;
    *(v8bf*)(&Bs[0][lofs0]) = sb0;
    *(v8bf*)(&Bs[0][lofs1]) = sb1;

    int buf = 0;
    for (int k0 = 0; k0 < K; k0 += 32) {
        __syncthreads();
        const bool more = (k0 + 32) < K;
        if (more) {               // issue next tile's global loads early
            gA0 += 32; gA1 += 32; gB0 += 32; gB1 += 32;
            sa0 = *(const v8bf*)gA0;
            sa1 = *(const v8bf*)gA1;
            sb0 = *(const v8bf*)gB0;
            sb1 = *(const v8bf*)gB1;
        }

        // all fragment LDS loads before the WMMA burst
        const bf16* Ab = &As[buf][0];
        const bf16* Bb = &Bs[buf][0];
        v16bf af[2], bfr[4];
#pragma unroll
        for (int i = 0; i < 2; ++i)
            af[i] = make_afrag(Ab + (wr + i * 16 + l15) * 32 + hi * 16);
#pragma unroll
        for (int t = 0; t < 4; ++t)
            bfr[t] = *(const v16bf*)(Bb + (wc + t * 16 + l15) * 32 + hi * 16);

#pragma unroll
        for (int i = 0; i < 2; ++i)
#pragma unroll
            for (int t = 0; t < 4; ++t)
                acc[i][t] = __builtin_amdgcn_wmma_f32_16x16x32_bf16(
                    false, af[i], false, bfr[t], (short)0, acc[i][t], false, false);

        if (more) {               // stash next tile into the other buffer
            const int nb = buf ^ 1;
            *(v8bf*)(&As[nb][lofs0]) = sa0;
            *(v8bf*)(&As[nb][lofs1]) = sa1;
            *(v8bf*)(&Bs[nb][lofs0]) = sb0;
            *(v8bf*)(&Bs[nb][lofs1]) = sb1;
        }
        buf ^= 1;
    }

    // epilogue: C/D layout -> row = 8*hi + r, col = lane&15
#pragma unroll
    for (int i = 0; i < 2; ++i) {
        const int rowb = m0 + wr + i * 16 + hi * 8;
#pragma unroll
        for (int t = 0; t < 4; ++t) {
            const int colg = n0 + wc + t * 16 + l15;
#pragma unroll
            for (int r = 0; r < 8; ++r)
                C[(ull)(rowb + r) * ldc + colg] = acc[i][t][r] * scale;
        }
    }
}

// ---------------------------------------------------------------------------
// Projection GEMM: q/k/v = x @ W^T; fp32 inputs converted to bf16 in registers.
// grid.z: 0 -> q, 1 -> k, 2 -> v (v written transposed per batch: vT[d][s])
// Same 128x128 / 8-wave / double-buffered pipeline.
// ---------------------------------------------------------------------------
__global__ __launch_bounds__(256) void wmma_proj(
    const float* __restrict__ X,
    const float* __restrict__ Wq,
    const float* __restrict__ Wk,
    const float* __restrict__ Wv,
    bf16* __restrict__ qb, bf16* __restrict__ kb, bf16* __restrict__ vT)
{
    __shared__ __attribute__((aligned(64))) bf16 As[2][128 * 32];
    __shared__ __attribute__((aligned(64))) bf16 Bs[2][128 * 32];

    const int tid  = threadIdx.x;
    const int wave = tid >> 5;
    const int lane = tid & 31;
    const int l15  = lane & 15;
    const int hi   = lane >> 4;
    const int wr   = (wave & 3) * 32;
    const int wc   = (wave >> 2) * 64;
    const int m0   = blockIdx.x * 128;   // over 8192 flattened (b,s) rows
    const int n0   = blockIdx.y * 128;   // over 1024 output features
    const int z    = blockIdx.z;
    const float* W = (z == 0) ? Wq : (z == 1) ? Wk : Wv;

    const int c0 = tid, c1 = tid + 256;
    const int ra0 = c0 >> 2, ka0 = (c0 & 3) << 3;
    const int ra1 = c1 >> 2, ka1 = (c1 & 3) << 3;
    const int lofs0 = ra0 * 32 + ka0;
    const int lofs1 = ra1 * 32 + ka1;
    const float* gA0 = X + (ull)(m0 + ra0) * MDIM + ka0;
    const float* gA1 = X + (ull)(m0 + ra1) * MDIM + ka1;
    const float* gB0 = W + (ull)(n0 + ra0) * MDIM + ka0;
    const float* gB1 = W + (ull)(n0 + ra1) * MDIM + ka1;

    const v8f vzero = {0.f,0.f,0.f,0.f,0.f,0.f,0.f,0.f};
    v8f acc[2][4];
#pragma unroll
    for (int i = 0; i < 2; ++i)
#pragma unroll
        for (int t = 0; t < 4; ++t) acc[i][t] = vzero;

    v8bf sa0 = cvt8(gA0);
    v8bf sa1 = cvt8(gA1);
    v8bf sb0 = cvt8(gB0);
    v8bf sb1 = cvt8(gB1);
    *(v8bf*)(&As[0][lofs0]) = sa0;
    *(v8bf*)(&As[0][lofs1]) = sa1;
    *(v8bf*)(&Bs[0][lofs0]) = sb0;
    *(v8bf*)(&Bs[0][lofs1]) = sb1;

    int buf = 0;
    for (int k0 = 0; k0 < MDIM; k0 += 32) {
        __syncthreads();
        const bool more = (k0 + 32) < MDIM;
        if (more) {
            gA0 += 32; gA1 += 32; gB0 += 32; gB1 += 32;
            sa0 = cvt8(gA0);
            sa1 = cvt8(gA1);
            sb0 = cvt8(gB0);
            sb1 = cvt8(gB1);
        }

        const bf16* Ab = &As[buf][0];
        const bf16* Bb = &Bs[buf][0];
        v16bf af[2], bfr[4];
#pragma unroll
        for (int i = 0; i < 2; ++i)
            af[i] = make_afrag(Ab + (wr + i * 16 + l15) * 32 + hi * 16);
#pragma unroll
        for (int t = 0; t < 4; ++t)
            bfr[t] = *(const v16bf*)(Bb + (wc + t * 16 + l15) * 32 + hi * 16);

#pragma unroll
        for (int i = 0; i < 2; ++i)
#pragma unroll
            for (int t = 0; t < 4; ++t)
                acc[i][t] = __builtin_amdgcn_wmma_f32_16x16x32_bf16(
                    false, af[i], false, bfr[t], (short)0, acc[i][t], false, false);

        if (more) {
            const int nb = buf ^ 1;
            *(v8bf*)(&As[nb][lofs0]) = sa0;
            *(v8bf*)(&As[nb][lofs1]) = sa1;
            *(v8bf*)(&Bs[nb][lofs0]) = sb0;
            *(v8bf*)(&Bs[nb][lofs1]) = sb1;
        }
        buf ^= 1;
    }

    if (z < 2) {
        bf16* out = (z == 0) ? qb : kb;
#pragma unroll
        for (int i = 0; i < 2; ++i) {
            const int rowb = m0 + wr + i * 16 + hi * 8;
#pragma unroll
            for (int t = 0; t < 4; ++t) {
                const int colg = n0 + wc + t * 16 + l15;
#pragma unroll
                for (int r = 0; r < 8; ++r)
                    out[(ull)(rowb + r) * MDIM + colg] = (bf16)acc[i][t][r];
            }
        }
    } else {
        // v transposed per batch: vT[b][d][s]
#pragma unroll
        for (int i = 0; i < 2; ++i) {
            const int rowb = m0 + wr + i * 16 + hi * 8;
#pragma unroll
            for (int t = 0; t < 4; ++t) {
                const int colg = n0 + wc + t * 16 + l15;
#pragma unroll
                for (int r = 0; r < 8; ++r) {
                    const int rg = rowb + r;
                    const int bb = rg >> 11;      // batch
                    const int sl = rg & 2047;     // seq position
                    vT[(ull)bb * (MDIM * SEQL) + (ull)colg * SEQL + sl] =
                        (bf16)acc[i][t][r];
                }
            }
        }
    }
}

// ---------------------------------------------------------------------------
// Row softmax over 2048 fp32 scores -> bf16 probabilities. One block per row.
// ---------------------------------------------------------------------------
__global__ __launch_bounds__(256) void softmax_row(
    const float* __restrict__ S, bf16* __restrict__ P)
{
    const int row = blockIdx.x;
    const int tid = threadIdx.x;
    const float* s = S + (ull)row * SEQL;
    bf16* p = P + (ull)row * SEQL;

    float v[8];
    float m = -3.402823466e38f;
#pragma unroll
    for (int i = 0; i < 8; ++i) { v[i] = s[tid + i * 256]; m = fmaxf(m, v[i]); }

    __shared__ float red[256];
    red[tid] = m;
    __syncthreads();
    for (int off = 128; off > 0; off >>= 1) {
        if (tid < off) red[tid] = fmaxf(red[tid], red[tid + off]);
        __syncthreads();
    }
    m = red[0];
    __syncthreads();

    float sum = 0.f;
#pragma unroll
    for (int i = 0; i < 8; ++i) { v[i] = __expf(v[i] - m); sum += v[i]; }
    red[tid] = sum;
    __syncthreads();
    for (int off = 128; off > 0; off >>= 1) {
        if (tid < off) red[tid] += red[tid + off];
        __syncthreads();
    }
    const float inv = 1.0f / red[0];
#pragma unroll
    for (int i = 0; i < 8; ++i) p[tid + i * 256] = (bf16)(v[i] * inv);
}

// ---------------------------------------------------------------------------
// Workspace layout (bytes):
//   [0,   16M)  qb  bf16 [4,2048,1024]   (later reused for probs)
//   [16M, 32M)  kb  bf16 [4,2048,1024]   (later reused for probs)
//   [32M, 48M)  vT  bf16 [4,1024,2048]
//   [48M,112M)  scores fp32 [4,2048,2048]
//   probs bf16 [4,2048,2048] aliases [0,32M) -- q/k are dead by then.
// ---------------------------------------------------------------------------
extern "C" void kernel_launch(void* const* d_in, const int* in_sizes, int n_in,
                              void* d_out, int out_size, void* d_ws, size_t ws_size,
                              hipStream_t stream) {
    (void)in_sizes; (void)n_in; (void)out_size; (void)ws_size;
    const float* x  = (const float*)d_in[0];
    const float* Wq = (const float*)d_in[1];
    const float* Wk = (const float*)d_in[2];
    const float* Wv = (const float*)d_in[3];
    float* out = (float*)d_out;

    char* ws = (char*)d_ws;
    bf16*  qb     = (bf16*)(ws);
    bf16*  kb     = (bf16*)(ws + (size_t)16 * 1024 * 1024);
    bf16*  vT     = (bf16*)(ws + (size_t)32 * 1024 * 1024);
    float* scores = (float*)(ws + (size_t)48 * 1024 * 1024);
    bf16*  probs  = (bf16*)(ws);   // reuses q+k space

    dim3 blk(256);

    // q,k,v projections (z selects weight / destination)
    wmma_proj<<<dim3(64, 8, 3), blk, 0, stream>>>(x, Wq, Wk, Wv, qb, kb, vT);

    // scores = (1/sqrt(1024)) * q @ k^T   [per batch 2048x2048]
    wmma_gemm_bf16<<<dim3(16, 16, NBATCH), blk, 0, stream>>>(
        qb, MDIM, (ull)SEQL * MDIM,
        kb, MDIM, (ull)SEQL * MDIM,
        scores, SEQL, (ull)SEQL * SEQL,
        MDIM, 0.03125f);

    // softmax over each of the 8192 rows
    softmax_row<<<dim3(NBATCH * SEQL), dim3(256), 0, stream>>>(scores, probs);

    // out = P @ v   (B columns are contiguous rows of vT)
    wmma_gemm_bf16<<<dim3(16, 8, NBATCH), blk, 0, stream>>>(
        probs, SEQL, (ull)SEQL * SEQL,
        vT, SEQL, (ull)MDIM * SEQL,
        out, MDIM, (ull)SEQL * MDIM,
        SEQL, 1.0f);
}
